// EncoderDecoder_4784593568121
// MI455X (gfx1250) — compile-verified, matrix operand order checked by reading, not run on hardware
//
#include <hip/hip_runtime.h>
#include <hip/hip_bf16.h>

// ---------------- types ----------------
typedef __attribute__((ext_vector_type(8)))  _Float16 v8h;
typedef __attribute__((ext_vector_type(16))) _Float16 v16h;
typedef __attribute__((ext_vector_type(8)))  float    v8f;

#define NT 4  // N-tiles per wave in the WMMA GEMM

__device__ __forceinline__ v16h load_frag16(const _Float16* p) {
    // two 16B loads -> one 16-element f16 fragment (matches 16-bit A/B striping)
    v8h lo = *(const v8h*)(p);
    v8h hi = *(const v8h*)(p + 16);
    return __builtin_shufflevector(lo, hi, 0,1,2,3,4,5,6,7,8,9,10,11,12,13,14,15);
}

__device__ __forceinline__ float sigf(float x) { return 1.f / (1.f + __expf(-x)); }

// ---------------- WMMA GEMM: C(M,N) = A(M,K) * W(N,K)^T + bias, optional relu ----------------
// A: f16 row-major (lda), W: f16 row-major (N,K) (ldw). M assumed <= 64, 4 waves/block.
__global__ __launch_bounds__(128)
void k_gemm_h16(const _Float16* __restrict__ A, int lda,
                const _Float16* __restrict__ W, int ldw,
                const float* __restrict__ bias,
                float* __restrict__ outF, int ldoF,
                _Float16* __restrict__ outH, int ldoH,
                int M, int N, int K, int relu)
{
    const int lane = threadIdx.x & 31;
    const int wave = threadIdx.x >> 5;
    const int mt = wave;
    if (mt * 16 >= M) return;
    const int Ntiles = N >> 4;
    const int nt0 = blockIdx.x * NT;
    if (nt0 >= Ntiles) return;

    const int row  = lane & 15;          // A row within tile / B column within tile
    const int koff = (lane >> 4) << 3;   // K-half select per ISA striping

    int ntile[NT];
    const _Float16* wrow[NT];
    v8f acc[NT] = {};
#pragma unroll
    for (int j = 0; j < NT; ++j) {
        int tIdx = nt0 + j;
        if (tIdx > Ntiles - 1) tIdx = Ntiles - 1;   // clamp tail tiles (store guarded below)
        ntile[j] = tIdx;
        wrow[j]  = W + (size_t)(tIdx * 16 + row) * (size_t)ldw;
    }
    const _Float16* arow = A + (size_t)(mt * 16 + row) * (size_t)lda;

    for (int k0 = 0; k0 < K; k0 += 32) {
        v16h a = load_frag16(arow + k0 + koff);
        __builtin_prefetch(arow + k0 + 64, 0, 0);
#pragma unroll
        for (int j = 0; j < NT; ++j) {
            v16h b = load_frag16(wrow[j] + k0 + koff);
            acc[j] = __builtin_amdgcn_wmma_f32_16x16x32_f16(
                         false, a, false, b, (short)0, acc[j], false, false);
        }
    }

    const int mbase = mt * 16 + ((lane >> 4) << 3);  // D layout: lanes>=16 hold M=8..15
    const int ncol  = lane & 15;
#pragma unroll
    for (int j = 0; j < NT; ++j) {
        if (nt0 + j >= Ntiles) break;
        int gn = ntile[j] * 16 + ncol;
        float bb = bias ? bias[gn] : 0.f;
#pragma unroll
        for (int r = 0; r < 8; ++r) {
            float v = acc[j][r] + bb;
            if (relu) v = fmaxf(v, 0.f);
            if (outF) outF[(size_t)(mbase + r) * (size_t)ldoF + gn] = v;
            if (outH) outH[(size_t)(mbase + r) * (size_t)ldoH + gn] = (_Float16)v;
        }
    }
}

// ---------------- prologue kernels ----------------
__global__ void k_f2h(const float* __restrict__ in, _Float16* __restrict__ out, int n) {
    int i = blockIdx.x * blockDim.x + threadIdx.x;
    if (i < n) out[i] = (_Float16)in[i];
}

__global__ void k_pack_lstm_w(const float* __restrict__ wih, const float* __restrict__ whh,
                              _Float16* __restrict__ wc, int N, int Kh) {
    int i = blockIdx.x * blockDim.x + threadIdx.x;
    int tot = N * 2 * Kh;
    if (i >= tot) return;
    int n = i / (2 * Kh), c = i % (2 * Kh);
    float v = (c < Kh) ? wih[(size_t)n * Kh + c] : whh[(size_t)n * Kh + (c - Kh)];
    wc[i] = (_Float16)v;
}

__global__ void k_bias_sum(const float* a, const float* b, float* o, int n) {
    int i = blockIdx.x * blockDim.x + threadIdx.x;
    if (i < n) o[i] = a[i] + b[i];
}

// mean over K of iml (B,H,K); one wave per (b,h) row -> f16 output
__global__ void k_mean(const float* __restrict__ iml, _Float16* __restrict__ mh, int BH, int K) {
    int w = (blockIdx.x * blockDim.x + threadIdx.x) >> 5;
    int lane = threadIdx.x & 31;
    if (w >= BH) return;
    const float* rowp = iml + (size_t)w * K;
    float s = 0.f;
    for (int k = lane; k < K; k += 32) s += rowp[k];
    for (int o = 16; o > 0; o >>= 1) s += __shfl_xor(s, o, 32);
    if (lane == 0) mh[w] = (_Float16)(s / (float)K);
}

// e_ann[b,k] = sum_h iml[b,h,k]*w_a[h] + attn_b  (coalesced over k)
__global__ void k_eann(const float* __restrict__ iml, const float* __restrict__ attn_w,
                       const float* __restrict__ attn_b, float* __restrict__ e_ann,
                       int B, int H, int K) {
    int i = blockIdx.x * blockDim.x + threadIdx.x;
    if (i >= B * K) return;
    int b = i / K, k = i % K;
    const float* base = iml + (size_t)b * H * K + k;
    float s = 0.f;
    for (int h = 0; h < H; ++h) s += base[(size_t)h * K] * attn_w[h];
    e_ann[i] = s + attn_b[0];
}

__global__ void k_pack_x0(const float* __restrict__ x, _Float16* __restrict__ xh,
                          int B, int E, int ld) {
    int i = blockIdx.x * blockDim.x + threadIdx.x;
    if (i >= B * E) return;
    int b = i / E, e = i % E;
    xh[(size_t)b * ld + e] = (_Float16)x[i];
}

// ---------------- per-step kernels ----------------
// gates g(B,4H) -> update c,h; write h f16 into xh[:,H:2H] and hc[:,0:H]
__global__ void k_lstm(const float* __restrict__ g, float* __restrict__ c, float* __restrict__ h,
                       _Float16* __restrict__ xh, _Float16* __restrict__ hc, int B, int H) {
    int i = blockIdx.x * blockDim.x + threadIdx.x;
    if (i >= B * H) return;
    int b = i / H, j = i % H;
    const float* gr = g + (size_t)b * 4 * H;
    float gi = gr[j], gf = gr[H + j], gg = gr[2 * H + j], go = gr[3 * H + j];
    float cn = sigf(gf) * c[i] + sigf(gi) * tanhf(gg);
    float hn = sigf(go) * tanhf(cn);
    c[i] = cn; h[i] = hn;
    xh[(size_t)b * 2 * H + H + j] = (_Float16)hn;
    hc[(size_t)b * 2 * H + j]     = (_Float16)hn;
}

// attention: block per batch. dot = h . w_h ; softmax over K ; cs[h] = sum_k alpha[k]*iml[b,h,k]
__global__ __launch_bounds__(256)
void k_attention(const float* __restrict__ h, const float* __restrict__ attn_w,
                 const float* __restrict__ e_ann, const float* __restrict__ iml,
                 _Float16* __restrict__ hc, int K, int H) {
    __shared__ float red[256];
    __shared__ float alpha[256];
    int b = blockIdx.x, t = threadIdx.x;
    // h . w_h  (w_h = attn_w[H:])
    float s = 0.f;
    for (int i = t; i < H; i += 256) s += h[(size_t)b * H + i] * attn_w[H + i];
    red[t] = s; __syncthreads();
    for (int o = 128; o > 0; o >>= 1) { if (t < o) red[t] += red[t + o]; __syncthreads(); }
    float dot = red[0]; __syncthreads();
    // softmax over K (K <= 256)
    float e = (t < K) ? (e_ann[(size_t)b * K + t] + dot) : -3.4e38f;
    red[t] = e; __syncthreads();
    for (int o = 128; o > 0; o >>= 1) { if (t < o) red[t] = fmaxf(red[t], red[t + o]); __syncthreads(); }
    float mx = red[0]; __syncthreads();
    float ex = (t < K) ? __expf(e - mx) : 0.f;
    red[t] = ex; __syncthreads();
    for (int o = 128; o > 0; o >>= 1) { if (t < o) red[t] += red[t + o]; __syncthreads(); }
    float inv = 1.f / red[0]; __syncthreads();
    alpha[t] = ex * inv; __syncthreads();
    // context: per-h dot over K (iml rows are contiguous in k)
    for (int hh = t; hh < H; hh += 256) {
        const float* base = iml + ((size_t)b * H + hh) * K;
        float cs = 0.f;
        for (int k = 0; k < K; ++k) cs += alpha[k] * base[k];
        hc[(size_t)b * 2 * H + H + hh] = (_Float16)cs;
    }
}

// argmax over V, first-max tie-break; writes int sample (ws) and float sample (d_out)
__global__ __launch_bounds__(256)
void k_argmax(const float* __restrict__ logits, int rowStride, int colOff, int V,
              int* __restrict__ samp, float* __restrict__ outS, int T, int tstep) {
    __shared__ float bv[256];
    __shared__ int   bi_s[256];
    int b = blockIdx.x, t = threadIdx.x;
    const float* row = logits + (size_t)b * rowStride + colOff;
    float best = -3.4e38f; int bi = 0;
    for (int v = t; v < V; v += 256) { float x = row[v]; if (x > best) { best = x; bi = v; } }
    bv[t] = best; bi_s[t] = bi; __syncthreads();
    for (int o = 128; o > 0; o >>= 1) {
        if (t < o) {
            if (bv[t + o] > bv[t] || (bv[t + o] == bv[t] && bi_s[t + o] < bi_s[t])) {
                bv[t] = bv[t + o]; bi_s[t] = bi_s[t + o];
            }
        }
        __syncthreads();
    }
    if (t == 0) { samp[b] = bi_s[0]; outS[(size_t)b * T + tstep] = (float)bi_s[0]; }
}

__global__ void k_embed(const float* __restrict__ emb, const int* __restrict__ samp,
                        _Float16* __restrict__ xh, int B, int E, int ld) {
    int i = blockIdx.x * blockDim.x + threadIdx.x;
    if (i >= B * E) return;
    int b = i / E, e = i % E;
    xh[(size_t)b * ld + e] = (_Float16)emb[(size_t)samp[b] * E + e];
}

// ---------------- host ----------------
extern "C" void kernel_launch(void* const* d_in, const int* in_sizes, int n_in,
                              void* d_out, int out_size, void* d_ws, size_t ws_size,
                              hipStream_t stream) {
    const float* iml      = (const float*)d_in[0];
    const float* img_feat = (const float*)d_in[1];
    const float* m1w1 = (const float*)d_in[2];
    const float* m1b1 = (const float*)d_in[3];
    const float* m1w2 = (const float*)d_in[4];
    const float* m1b2 = (const float*)d_in[5];
    const float* m2w1 = (const float*)d_in[6];
    const float* m2b1 = (const float*)d_in[7];
    const float* m2w2 = (const float*)d_in[8];
    const float* m2b2 = (const float*)d_in[9];
    const float* wih  = (const float*)d_in[10];
    const float* whh  = (const float*)d_in[11];
    const float* bih  = (const float*)d_in[12];
    const float* bhh  = (const float*)d_in[13];
    const float* attn_w = (const float*)d_in[14];
    const float* attn_b = (const float*)d_in[15];
    const float* fc_w = (const float*)d_in[16];
    const float* fc_b = (const float*)d_in[17];
    const float* emb  = (const float*)d_in[18];

    // derive dims from sizes
    const int H = in_sizes[3];                 // 512
    const int V = in_sizes[17];                // 10000
    const int E = in_sizes[18] / V;            // 512
    const int B = in_sizes[1] / E;             // 64
    const int K = in_sizes[0] / (B * H);       // 196
    const int T = out_size / (B * (V + 1));    // 30
    const int G4 = 4 * H;                      // 2048
    const int KC = E + H;                      // 1024 (concat contraction)

    // workspace carve-out
    char* ws = (char*)d_ws;
    size_t off = 0;
    auto alloc = [&](size_t bytes) { size_t r = off; off = (off + bytes + 255) & ~(size_t)255; return r; };
    _Float16* fcwh  = (_Float16*)(ws + alloc((size_t)V * KC * 2));
    _Float16* wcomb = (_Float16*)(ws + alloc((size_t)G4 * KC * 2));
    _Float16* w11h  = (_Float16*)(ws + alloc((size_t)H * H * 2));
    _Float16* w12h  = (_Float16*)(ws + alloc((size_t)H * H * 2));
    _Float16* w21h  = (_Float16*)(ws + alloc((size_t)H * H * 2));
    _Float16* w22h  = (_Float16*)(ws + alloc((size_t)H * H * 2));
    float*    bsum  = (float*)(ws + alloc((size_t)G4 * 4));
    _Float16* meanh = (_Float16*)(ws + alloc((size_t)B * H * 2));
    _Float16* t1h   = (_Float16*)(ws + alloc((size_t)B * H * 2));
    float*    hbuf  = (float*)(ws + alloc((size_t)B * H * 4));
    float*    cbuf  = (float*)(ws + alloc((size_t)B * H * 4));
    _Float16* xh    = (_Float16*)(ws + alloc((size_t)B * KC * 2));
    _Float16* hc    = (_Float16*)(ws + alloc((size_t)B * KC * 2));
    float*    gbuf  = (float*)(ws + alloc((size_t)B * G4 * 4));
    float*    e_ann = (float*)(ws + alloc((size_t)B * K * 4));
    int*      samp  = (int*)(ws + alloc((size_t)B * 4));
    (void)ws_size; (void)n_in;

    auto grid1 = [](int n) { return dim3((unsigned)((n + 255) / 256)); };

    // -------- prologue: weight conversion / packing --------
    k_f2h<<<grid1(V * KC), 256, 0, stream>>>(fc_w, fcwh, V * KC);
    k_pack_lstm_w<<<grid1(G4 * KC), 256, 0, stream>>>(wih, whh, wcomb, G4, H);
    k_f2h<<<grid1(H * H), 256, 0, stream>>>(m1w1, w11h, H * H);
    k_f2h<<<grid1(H * H), 256, 0, stream>>>(m1w2, w12h, H * H);
    k_f2h<<<grid1(H * H), 256, 0, stream>>>(m2w1, w21h, H * H);
    k_f2h<<<grid1(H * H), 256, 0, stream>>>(m2w2, w22h, H * H);
    k_bias_sum<<<grid1(G4), 256, 0, stream>>>(bih, bhh, bsum, G4);
    k_mean<<<grid1(B * H * 32), 256, 0, stream>>>(iml, meanh, B * H, K);
    k_eann<<<grid1(B * K), 256, 0, stream>>>(iml, attn_w, attn_b, e_ann, B, H, K);
    k_pack_x0<<<grid1(B * E), 256, 0, stream>>>(img_feat, xh, B, E, KC);

    auto gemm = [&](const _Float16* A, int lda, const _Float16* W, int ldw, const float* bias,
                    float* oF, int ldoF, _Float16* oH, int ldoH, int N, int Kk, int relu) {
        int ntiles = N / 16;
        dim3 g((unsigned)((ntiles + NT - 1) / NT));
        k_gemm_h16<<<g, 128, 0, stream>>>(A, lda, W, ldw, bias, oF, ldoF, oH, ldoH, B, N, Kk, relu);
    };

    // -------- h0 = MLP1(mean), c0 = MLP2(mean) --------
    gemm(meanh, H, w11h, H, m1b1, nullptr, 0, t1h, H, H, H, 1);
    gemm(t1h,   H, w12h, H, m1b2, hbuf, H, xh + H, KC, H, H, 1);   // h0 -> f32 state + concat f16
    gemm(meanh, H, w21h, H, m2b1, nullptr, 0, t1h, H, H, H, 1);
    gemm(t1h,   H, w22h, H, m2b2, cbuf, H, nullptr, 0, H, H, 1);   // c0 -> f32 state

    float* logits = (float*)d_out;                  // (B, T, V)
    float* outS   = (float*)d_out + (size_t)B * T * V;  // (B, T) samples as float

    // -------- decode loop --------
    for (int t = 0; t < T; ++t) {
        // gates = [x|h] @ [wih|whh]^T + (bih+bhh)
        gemm(xh, KC, wcomb, KC, bsum, gbuf, G4, nullptr, 0, G4, KC, 0);
        k_lstm<<<grid1(B * H), 256, 0, stream>>>(gbuf, cbuf, hbuf, xh, hc, B, H);
        k_attention<<<dim3((unsigned)B), 256, 0, stream>>>(hbuf, attn_w, e_ann, iml, hc, K, H);
        // logits written straight into d_out at (b, t, :) via ldc = T*V
        gemm(hc, KC, fcwh, KC, fc_b, logits + (size_t)t * V, T * V, nullptr, 0, V, KC, 0);
        k_argmax<<<dim3((unsigned)B), 256, 0, stream>>>(logits, T * V, t * V, V, samp, outS, T, t);
        k_embed<<<grid1(B * E), 256, 0, stream>>>(emb, samp, xh, B, E, KC);
    }
}
